// EffEnsembleDynamicModel_71708773974359
// MI455X (gfx1250) — compile-verified, matrix operand order checked by reading, not run on hardware
//
#include <hip/hip_runtime.h>

typedef __attribute__((ext_vector_type(16))) __bf16 v16bf;
typedef __attribute__((ext_vector_type(8)))  float  v8f;

namespace {
constexpr int kObs = 32, kAct = 8, kN = 100000, kE = 5, kM = 7;
constexpr int kHid = 200, kCO = 33, kC = 40;
constexpr int kR = (kN - 1) / kE + 1;          // 20000
constexpr int kRowsPerBlk = 64;                // 4 waves x 16 rows
constexpr int kNT_H = 13;                      // 208 cols (200 padded)
constexpr int kKT_H = 7;                       // 224 K (200 padded)
constexpr int kNT_O = 5;                       // 80 cols (66 padded)
constexpr int kKT_Z = 2;                       // 64 K (40 padded)
constexpr int kLdZ = 72;                       // bf16 row stride for z buffer
constexpr int kLdH = 232;                      // bf16 row stride for hidden buffers
constexpr int kLdO = 84;                       // f32 row stride for output buffer
constexpr size_t kFragElems = 512;             // 32x16 bf16 per fragment
constexpr size_t kOffL0 = 0;
constexpr size_t kOffL1 = kOffL0 + (size_t)kKT_Z * kNT_H * kFragElems;  // 13312
constexpr size_t kOffL2 = kOffL1 + (size_t)kKT_H * kNT_H * kFragElems;  // 59904
constexpr size_t kOffL3 = kOffL2 + (size_t)kKT_H * kNT_H * kFragElems;  // 106496
constexpr size_t kPerModel = kOffL3 + (size_t)kKT_H * kNT_O * kFragElems; // 124416 bf16
constexpr int kFragsPerModel = kKT_Z * kNT_H + 2 * kKT_H * kNT_H + kKT_H * kNT_O; // 243
}

union FragU { uint4 q[2]; v16bf v; };

__device__ __forceinline__ float softplusf(float x) {
  // numerically-stable log(1 + e^x)
  return (x > 0.0f) ? (x + log1pf(__expf(-x))) : log1pf(__expf(x));
}

// ---------------------------------------------------------------------------
// Prep: repack f32 weights into bf16 WMMA B-fragments (32x16 per fragment).
// Fragment dword d = lane*8 + j holds packed bf16 pair (K0, K0+1), with
//   lanes 0..15 : N = nt*16 + lane, K pairs covering K = 0..7, 16..23 of the chunk
//   lanes 16..31: N = nt*16 + lane-16, K pairs covering K = 8..15, 24..31
// matching the CDNA5 16-bit B-matrix 32x16 VGPR striping.
// ---------------------------------------------------------------------------
__global__ __launch_bounds__(256) void repack_weights_kernel(
    const float* __restrict__ W0, const float* __restrict__ W1,
    const float* __restrict__ W2, const float* __restrict__ W3,
    __bf16* __restrict__ ws) {
  int f = blockIdx.x;
  const int m = f / kFragsPerModel;
  f -= m * kFragsPerModel;
  const float* W;
  int Kdim, Ndim, NT, ldw, kt, nt;
  size_t base = (size_t)m * kPerModel;
  if (f < kKT_Z * kNT_H) {
    kt = f / kNT_H; nt = f % kNT_H;
    W = W0 + (size_t)m * kC * kHid; Kdim = kC; Ndim = kHid; NT = kNT_H; ldw = kHid;
    base += kOffL0;
  } else if (f < kKT_Z * kNT_H + kKT_H * kNT_H) {
    f -= kKT_Z * kNT_H;
    kt = f / kNT_H; nt = f % kNT_H;
    W = W1 + (size_t)m * kHid * kHid; Kdim = kHid; Ndim = kHid; NT = kNT_H; ldw = kHid;
    base += kOffL1;
  } else if (f < kKT_Z * kNT_H + 2 * kKT_H * kNT_H) {
    f -= kKT_Z * kNT_H + kKT_H * kNT_H;
    kt = f / kNT_H; nt = f % kNT_H;
    W = W2 + (size_t)m * kHid * kHid; Kdim = kHid; Ndim = kHid; NT = kNT_H; ldw = kHid;
    base += kOffL2;
  } else {
    f -= kKT_Z * kNT_H + 2 * kKT_H * kNT_H;
    kt = f / kNT_O; nt = f % kNT_O;
    W = W3 + (size_t)m * kHid * 2 * kCO; Kdim = kHid; Ndim = 2 * kCO; NT = kNT_O; ldw = 2 * kCO;
    base += kOffL3;
  }
  const int d = threadIdx.x;            // dword index within fragment, 0..255
  const int lane = d >> 3, j = d & 7;
  const int khalf = (lane >= 16) ? 8 : 0;
  const int k0 = kt * 32 + ((j < 4) ? (khalf + 2 * j) : (16 + khalf + 2 * (j - 4)));
  const int nc = nt * 16 + (lane & 15);
  const float w0v = (k0 < Kdim && nc < Ndim) ? W[(size_t)k0 * ldw + nc] : 0.0f;
  const float w1v = (k0 + 1 < Kdim && nc < Ndim) ? W[(size_t)(k0 + 1) * ldw + nc] : 0.0f;
  union { __bf16 h[2]; unsigned u; } p;
  p.h[0] = (__bf16)w0v;
  p.h[1] = (__bf16)w1v;
  ((unsigned*)(ws + base + ((size_t)(kt * NT + nt) << 9)))[d] = p.u;
}

// ---------------------------------------------------------------------------
// One GEMM layer for a wave's 16-row strip: LDS activations x L2-resident
// weight fragments via v_wmma_f32_16x16x32_bf16, bias + optional SiLU,
// written back to LDS (bf16 for hidden layers, f32 for the head).
// ---------------------------------------------------------------------------
template <int KT, int NT, bool SILU, bool F32OUT>
__device__ __forceinline__ void layer_gemm(const __bf16* in, int ldin,
                                           const __bf16* __restrict__ wfrag,
                                           const float* __restrict__ bias, int ndim,
                                           __bf16* outb, float* outf, int ldo,
                                           int wrow, int lane) {
  const int mrow  = lane & 15;
  const int khalf = (lane >> 4) << 3;           // 0 or 8 (A-layout K half)
  const int rbase = wrow + ((lane >> 4) << 3);  // D-layout row base
  const __bf16* arow = in + (size_t)(wrow + mrow) * ldin;
  __builtin_prefetch(wfrag, 0, 3);
  for (int nt = 0; nt < NT; ++nt) {
    v8f acc = {};
#pragma unroll
    for (int kt = 0; kt < KT; ++kt) {
      FragU a, b;
      a.q[0] = *(const uint4*)(arow + kt * 32 + khalf);        // K khalf..khalf+7
      a.q[1] = *(const uint4*)(arow + kt * 32 + 16 + khalf);   // K 16+khalf..
      const uint4* bp =
          (const uint4*)(wfrag + (((size_t)(kt * NT + nt)) << 9) + ((size_t)lane << 4));
      b.q[0] = bp[0];
      b.q[1] = bp[1];
      acc = __builtin_amdgcn_wmma_f32_16x16x32_bf16(
          false, a.v, false, b.v, (short)0, acc, false, false);
    }
    const int colg = nt * 16 + mrow;
    const float bv = (colg < ndim) ? bias[colg] : 0.0f;
#pragma unroll
    for (int v = 0; v < 8; ++v) {
      float x = acc[v] + bv;
      if (SILU) x = x / (1.0f + __expf(-x));
      if (F32OUT) {
        outf[(size_t)(rbase + v) * ldo + colg] = (colg < ndim) ? x : 0.0f;
      } else {
        outb[(size_t)(rbase + v) * ldo + colg] = (colg < ndim) ? (__bf16)x : (__bf16)0.0f;
      }
    }
  }
}

// ---------------------------------------------------------------------------
// Fused: gather -> normalize -> 4-layer MLP (bf16 WMMA) -> logvar clamps ->
// sample -> scatter next_obs / reward / terminal.
// ---------------------------------------------------------------------------
__global__ __launch_bounds__(128) void ensemble_mlp_kernel(
    const float* __restrict__ obs, const float* __restrict__ act,
    const float* __restrict__ scaler, const float* __restrict__ rscaler,
    const float* __restrict__ b0, const float* __restrict__ b1,
    const float* __restrict__ b2, const float* __restrict__ b3,
    const float* __restrict__ maxlv, const float* __restrict__ minlv,
    const float* __restrict__ noise, const int* __restrict__ idxs,
    const int* __restrict__ elites, const __bf16* __restrict__ wfrags,
    float* __restrict__ out) {
  __shared__ __align__(16) __bf16 hA[kRowsPerBlk * kLdH];
  __shared__ __align__(16) __bf16 hB[kRowsPerBlk * kLdH];
  __shared__ int srcbuf[kRowsPerBlk];
  __bf16* zbuf = hB;                 // z staging aliases hB (free until L1)
  float* outbuf = (float*)hB;        // head output aliases hB (free after L2)

  const int lane = threadIdx.x & 31;
  const int wave = threadIdx.x >> 5;
  const int wrow = wave * 16;
  const int r0 = blockIdx.x * kRowsPerBlk;
  const int e = blockIdx.y;
  const int me = elites[e];

  // per-row gather/scatter index (permutation of 0..N-1; same index both ways)
  if (lane < 16) {
    const int rloc = wrow + lane;
    const int rg = r0 + rloc;
    srcbuf[rloc] = (rg < kR) ? idxs[(size_t)e * kR + rg] : -1;
  }
  // stage normalized z as bf16 (K padded 40->64), zeros for invalid rows
  for (int t = lane; t < 16 * kLdZ; t += 32) {
    const int rl = t / kLdZ, c = t - rl * kLdZ;
    const int rloc = wrow + rl;
    const int src = srcbuf[rloc];
    float v = 0.0f;
    if (src >= 0 && c < kC) {
      const float raw = (c < kObs) ? obs[(size_t)src * kObs + c]
                                   : act[(size_t)src * kAct + (c - kObs)];
      v = (raw - scaler[c]) / scaler[kC + c];
    }
    zbuf[(size_t)rloc * kLdZ + c] = (__bf16)v;
  }
  // zero K-pad columns 208..231 of hA (never written by N-tiles)
  for (int t = lane; t < 16 * (kLdH - 208); t += 32) {
    const int rl = t / (kLdH - 208), c = 208 + t % (kLdH - 208);
    hA[(size_t)(wrow + rl) * kLdH + c] = (__bf16)0.0f;
  }
  __syncthreads();

  const __bf16* wm = wfrags + (size_t)me * kPerModel;

  // L0: z (64-K) -> hA
  layer_gemm<kKT_Z, kNT_H, true, false>(zbuf, kLdZ, wm + kOffL0,
                                        b0 + (size_t)me * kHid, kHid,
                                        hA, nullptr, kLdH, wrow, lane);
  __syncthreads();
  // zbuf no longer needed: zero hB's K-pad columns before it becomes a layer buffer
  for (int t = lane; t < 16 * (kLdH - 208); t += 32) {
    const int rl = t / (kLdH - 208), c = 208 + t % (kLdH - 208);
    hB[(size_t)(wrow + rl) * kLdH + c] = (__bf16)0.0f;
  }
  __syncthreads();
  // L1: hA -> hB
  layer_gemm<kKT_H, kNT_H, true, false>(hA, kLdH, wm + kOffL1,
                                        b1 + (size_t)me * kHid, kHid,
                                        hB, nullptr, kLdH, wrow, lane);
  __syncthreads();
  // L2: hB -> hA
  layer_gemm<kKT_H, kNT_H, true, false>(hB, kLdH, wm + kOffL2,
                                        b2 + (size_t)me * kHid, kHid,
                                        hA, nullptr, kLdH, wrow, lane);
  __syncthreads();
  // L3 (head): hA -> outbuf (f32, aliases hB)
  layer_gemm<kKT_H, kNT_O, false, true>(hA, kLdH, wm + kOffL3,
                                        b3 + (size_t)me * 2 * kCO, 2 * kCO,
                                        nullptr, outbuf, kLdO, wrow, lane);
  __syncthreads();

  // epilogue: logvar soft clamps, sample, scatter
  float* nextobs = out;
  float* reward = out + (size_t)kN * kObs;
  float* terminal = out + (size_t)kN * (kObs + 1);
  const float rs0 = rscaler[0], rs1 = rscaler[1];
  for (int t = lane; t < 16 * kCO; t += 32) {
    const int rl = t / kCO, c = t - rl * kCO;
    const int rloc = wrow + rl;
    const int rg = r0 + rloc;
    if (rg >= kR) continue;
    const int dst = srcbuf[rloc];
    const float mean = outbuf[(size_t)rloc * kLdO + c];
    float lv = outbuf[(size_t)rloc * kLdO + kCO + c];
    const float mx = maxlv[c], mn = minlv[c];
    lv = mx - softplusf(mx - lv);
    lv = mn + softplusf(lv - mn);
    const float s =
        mean + noise[((size_t)me * kR + rg) * kCO + c] * __expf(0.5f * lv);
    if (c < kObs) {
      nextobs[(size_t)dst * kObs + c] = s + obs[(size_t)dst * kObs + c];
    } else {
      reward[dst] = s * rs0 + rs1;
      terminal[dst] = 0.0f;
    }
  }
}

extern "C" void kernel_launch(void* const* d_in, const int* in_sizes, int n_in,
                              void* d_out, int out_size, void* d_ws, size_t ws_size,
                              hipStream_t stream) {
  (void)in_sizes; (void)n_in; (void)out_size; (void)ws_size;
  const float* obs = (const float*)d_in[0];
  const float* act = (const float*)d_in[1];
  const float* scaler = (const float*)d_in[2];
  const float* rscaler = (const float*)d_in[3];
  const float* W0 = (const float*)d_in[4];
  const float* b0 = (const float*)d_in[5];
  const float* W1 = (const float*)d_in[6];
  const float* b1 = (const float*)d_in[7];
  const float* W2 = (const float*)d_in[8];
  const float* b2 = (const float*)d_in[9];
  const float* W3 = (const float*)d_in[10];
  const float* b3 = (const float*)d_in[11];
  const float* maxlv = (const float*)d_in[12];
  const float* minlv = (const float*)d_in[13];
  const float* noise = (const float*)d_in[14];
  const int* idxs = (const int*)d_in[15];
  const int* elites = (const int*)d_in[16];
  __bf16* wfrags = (__bf16*)d_ws;   // needs kM * kPerModel * 2 = ~1.7 MB

  repack_weights_kernel<<<kM * kFragsPerModel, 256, 0, stream>>>(W0, W1, W2, W3,
                                                                 wfrags);
  dim3 grid((kR + kRowsPerBlk - 1) / kRowsPerBlk, kE);
  ensemble_mlp_kernel<<<grid, 128, 0, stream>>>(
      obs, act, scaler, rscaler, b0, b1, b2, b3, maxlv, minlv, noise, idxs,
      elites, wfrags, (float*)d_out);
}
